// TreeLSTMCell_68624987455860
// MI455X (gfx1250) — matrix operand order repaired and checked by reading, not compiled, not run on hardware
//
#include <hip/hip_runtime.h>
#include <hip/hip_bf16.h>
#include <math.h>

// ---------------- problem constants (match reference) ----------------
#define H        64
#define IN_SZ    4800
#define DEPTH    14
#define N_NODES  16383       // 2^14 - 1
#define M_PAD    16384       // padded row count for the big GEMM
#define FOURH    256         // 4*H
#define KSTEPS   150         // 4800 / 32
#define NTILES   16          // 256 / 16
#define NPB      4           // nodes per block in tree-level kernel

typedef __attribute__((ext_vector_type(16))) __bf16 bf16x16;
typedef __attribute__((ext_vector_type(8)))  float  v8f;

union BfVec { bf16x16 v; unsigned int u[8]; uint4 q[2]; };

__device__ __forceinline__ unsigned short f2bf(float f) {
  // round-to-nearest-even fp32 -> bf16
  unsigned int u = __float_as_uint(f);
  u += 0x7FFFu + ((u >> 16) & 1u);
  return (unsigned short)(u >> 16);
}

__device__ __forceinline__ float sigmoidf(float x) {
  return 1.0f / (1.0f + __expf(-x));
}

// ---------------------------------------------------------------------
// Kernel 1: convert Wx_w (fp32 [256,4800], row = K-contiguous) to bf16 and
// pre-swizzle into a per-lane WMMA B layout (contiguous k, same order as A):
//   B tile t (cols t*16..t*16+15), k-step s (K = s*32..s*32+31)
//   lane L: column n = t*16 + (L&15); K half = (L>>4)*16; VGPR j -> K = half+2j,2j+1
// Stored as [s][t][lane][8 dwords] so the GEMM loads 2x b128 per tile.
// ---------------------------------------------------------------------
__global__ void swizzleB(const float* __restrict__ Wxw,
                         unsigned int* __restrict__ bswz) {
  int tid = blockIdx.x * blockDim.x + threadIdx.x;
  if (tid >= KSTEPS * NTILES * 32 * 8) return;
  int j    = tid & 7;
  int lane = (tid >> 3) & 31;
  int t    = (tid >> 8) & 15;
  int s    = tid >> 12;
  int n = t * 16 + (lane & 15);
  int k = s * 32 + ((lane >> 4) << 4) + (j << 1);
  const float* row = Wxw + (size_t)n * IN_SZ + k;
  unsigned int lo = f2bf(row[0]);
  unsigned int hi = f2bf(row[1]);
  bswz[((size_t)(s * NTILES + t) * 32 + lane) * 8 + j] = (hi << 16) | lo;
}

// ---------------------------------------------------------------------
// Kernel 2: Wx = inputs @ Wx_w.T + Wx_b via v_wmma_f32_16x16x32_bf16.
// One wave per 16-row M-tile; each wave sweeps all 16 N-tiles so the fp32
// activations are read from HBM exactly once (bf16-converted in registers).
// Per k-step: issue ALL 32 B b128 loads (distinct dest regs -> one clause,
// incremental waits), overlap them with the A load+convert VALU work, then
// run the 16 WMMAs while later B tiles are still in flight.
// ---------------------------------------------------------------------
__global__ void __launch_bounds__(128)
gemm_wx(const float* __restrict__ X, const unsigned int* __restrict__ bswz,
        const float* __restrict__ bias, float* __restrict__ Wx) {
  const int lane   = threadIdx.x & 31;
  const int wave   = threadIdx.x >> 5;
  const int mt     = blockIdx.x * 4 + wave;     // 0..1023
  const int laneHi = lane >> 4;
  const int rIT    = lane & 15;
  const int m      = mt * 16 + rIT;             // A row loaded by this lane
  const bool mvalid = (m < N_NODES);
  const float* arow = X + (size_t)m * IN_SZ;

  v8f zero = {};
  v8f acc[NTILES];
#pragma unroll
  for (int t = 0; t < NTILES; ++t) acc[t] = zero;

  for (int s = 0; s < KSTEPS; ++s) {
    // ---- issue all B loads first (16 tiles x 2 b128, distinct registers)
    BfVec b[NTILES];
    const uint4* bq = (const uint4*)(bswz + ((size_t)s * 512 + lane) * 8);
#pragma unroll
    for (int t = 0; t < NTILES; ++t) {
      b[t].q[0] = bq[t * 64 + 0];
      b[t].q[1] = bq[t * 64 + 1];
    }

    // ---- A: 8 guarded float2 loads + bf16 convert (overlaps B latency)
    BfVec a;
#pragma unroll
    for (int j = 0; j < 8; ++j) {
      int k = s * 32 + (laneHi << 4) + (j << 1);
      float x0 = 0.f, x1 = 0.f;
      if (mvalid) { x0 = arow[k]; x1 = arow[k + 1]; }
      a.u[j] = ((unsigned int)f2bf(x1) << 16) | (unsigned int)f2bf(x0);
    }
    // prefetch the A stream ahead (A is the only HBM-resident stream)
    if (mvalid && s + 8 < KSTEPS) __builtin_prefetch(arow + (s + 8) * 32, 0, 3);

    // ---- 16 WMMAs; later B tiles still streaming in from L2
#pragma unroll
    for (int t = 0; t < NTILES; ++t) {
      acc[t] = __builtin_amdgcn_wmma_f32_16x16x32_bf16(
          /*neg_a=*/false, a.v, /*neg_b=*/false, b[t].v,
          /*c_mod=*/(short)0, acc[t], /*reuse_a=*/false, /*reuse_b=*/false);
    }
  }

  // C/D layout: VGPR v, lanes 0-15 -> (M=v, N=lane); lanes 16-31 -> (M=v+8).
  const int rbase = mt * 16 + laneHi * 8;       // < 16384, padded ws: no guard
#pragma unroll
  for (int t = 0; t < NTILES; ++t) {
    const int col = t * 16 + rIT;
    const float bv = bias[col];
#pragma unroll
    for (int v = 0; v < 8; ++v)
      Wx[(size_t)(rbase + v) * FOURH + col] = acc[t][v] + bv;
  }
}

// ---------------------------------------------------------------------
// Kernel 3: leaf-level gate biases  b_x[h] = dot(U_x[h,:], conv_b)
// (leaves have avg == conv_b, sum_f == 0).
// ---------------------------------------------------------------------
__global__ void leaf_bias(const float* __restrict__ Ui, const float* __restrict__ Uo,
                          const float* __restrict__ Uu, const float* __restrict__ cb,
                          float* __restrict__ bias3) {
  int idx = threadIdx.x;                        // 0..191
  if (idx >= 192) return;
  const float* U = (idx < 64) ? Ui : (idx < 128) ? Uo : Uu;
  int h = idx & 63;
  float s = 0.f;
  for (int m = 0; m < H; ++m) s += U[h * H + m] * cb[m];
  bias3[idx] = s;
}

// ---------------------------------------------------------------------
// Kernel 4: leaves (level d = 13, 8192 nodes). Pure elementwise gates.
// ---------------------------------------------------------------------
__global__ void leaves(const float* __restrict__ Wx, const float* __restrict__ bias3,
                       float* __restrict__ h_all, float* __restrict__ c_all) {
  int tid = blockIdx.x * blockDim.x + threadIdx.x;  // 8192*64 threads
  int h = tid & 63;
  int j = tid >> 6;
  int g = 8191 + j;                                  // global node id
  const float* w = Wx + (size_t)g * FOURH;
  float i = sigmoidf(w[h]       + bias3[h]);
  float o = sigmoidf(w[128 + h] + bias3[64 + h]);
  float u = tanhf   (w[192 + h] + bias3[128 + h]);
  float c = i * u;
  h_all[(size_t)g * H + h] = o * tanhf(c);
  c_all[(size_t)g * H + h] = c;
}

// ---------------------------------------------------------------------
// Kernel 5: one inner tree level. 64 threads per node (thread = output h),
// NPB nodes per block. Children of global node g are 2g+1, 2g+2.
// U matrices / convW are tiny (<=32 KB) and stay hot in WGP$/L2.
// ---------------------------------------------------------------------
__global__ void __launch_bounds__(64 * NPB)
tree_level(const float* __restrict__ Wx,
           const float* __restrict__ Ui, const float* __restrict__ Uf,
           const float* __restrict__ Uo, const float* __restrict__ Uu,
           const float* __restrict__ convW, const float* __restrict__ cb,
           float* __restrict__ h_all, float* __restrict__ c_all,
           int start, int n, float* __restrict__ out, int is_root) {
  __shared__ float sh[NPB][2 * H];
  __shared__ float sc[NPB][2 * H];
  __shared__ float sa[NPB][H];
  const int h = threadIdx.x;                 // 0..63
  const int y = threadIdx.y;                 // 0..NPB-1
  const int j = blockIdx.x * NPB + y;
  const bool active = (j < n);
  const int g = start + j;

  if (active) {
    const int c1 = 2 * g + 1, c2 = 2 * g + 2;
    sh[y][h]     = h_all[(size_t)c1 * H + h];
    sh[y][H + h] = h_all[(size_t)c2 * H + h];
    sc[y][h]     = c_all[(size_t)c1 * H + h];
    sc[y][H + h] = c_all[(size_t)c2 * H + h];
  }
  __syncthreads();

  if (active) {
    // avg[h] = conv_b[h] + sum_{k,m} convW[h,k,m] * ch_h[k,m]
    float avg = cb[h];
    const float* cw = convW + h * 2 * H;
    for (int kk = 0; kk < 2 * H; ++kk) avg += cw[kk] * sh[y][kk];
    sa[y][h] = avg;
  }
  __syncthreads();
  if (!active) return;

  const float* w = Wx + (size_t)g * FOURH;

  // forget gates (one per child) and sum_f
  float xf = w[64 + h];
  float a1 = 0.f, a2 = 0.f;
  const float* uf = Uf + h * H;
  for (int m = 0; m < H; ++m) {
    float wm = uf[m];
    a1 += wm * sh[y][m];
    a2 += wm * sh[y][H + m];
  }
  float f1 = sigmoidf(xf + a1);
  float f2 = sigmoidf(xf + a2);
  float sumf = f1 * sc[y][h] + f2 * sc[y][H + h];

  // i / o / u from avg
  float di = 0.f, doo = 0.f, du = 0.f;
  const float* ui = Ui + h * H;
  const float* uo = Uo + h * H;
  const float* uu = Uu + h * H;
  for (int m = 0; m < H; ++m) {
    float am = sa[y][m];
    di  += ui[m] * am;
    doo += uo[m] * am;
    du  += uu[m] * am;
  }
  float i = sigmoidf(w[h]       + di);
  float o = sigmoidf(w[128 + h] + doo);
  float u = tanhf   (w[192 + h] + du);
  float c = sumf + i * u;
  float hh = o * tanhf(c);
  h_all[(size_t)g * H + h] = hh;
  c_all[(size_t)g * H + h] = c;
  if (is_root) { out[h] = hh; out[H + h] = c; }
}

// ---------------------------------------------------------------------
extern "C" void kernel_launch(void* const* d_in, const int* in_sizes, int n_in,
                              void* d_out, int out_size, void* d_ws, size_t ws_size,
                              hipStream_t stream) {
  const float* X   = (const float*)d_in[0];   // [16383, 4800]
  const float* Wxw = (const float*)d_in[1];   // [256, 4800]
  const float* Wxb = (const float*)d_in[2];   // [256]
  const float* Ui  = (const float*)d_in[3];   // [64,64]
  const float* Uf  = (const float*)d_in[4];
  const float* Uo  = (const float*)d_in[5];
  const float* Uu  = (const float*)d_in[6];
  const float* cW  = (const float*)d_in[7];   // [64,2,64]
  const float* cb  = (const float*)d_in[8];   // [64]
  float* out = (float*)d_out;                 // 128 floats: h_root, c_root

  // workspace layout (bytes)
  char* ws = (char*)d_ws;
  float*        Wx    = (float*)(ws + 0);               // 16384*256*4 = 16,777,216
  unsigned int* bswz  = (unsigned int*)(ws + 16777216); // 2,457,600
  float*        h_all = (float*)(ws + 19234816);        // 16383*64*4 = 4,194,048
  float*        c_all = (float*)(ws + 23428864);        // 4,194,048
  float*        bias3 = (float*)(ws + 27622912);        // 768

  // 1) pre-swizzle weights into bf16 WMMA-B layout (tiny, L2-resident)
  swizzleB<<<(KSTEPS * NTILES * 32 * 8) / 256, 256, 0, stream>>>(Wxw, bswz);

  // 2) big GEMM: 1024 M-tiles, 4 waves/block
  gemm_wx<<<1024 / 4, 128, 0, stream>>>(X, bswz, Wxb, Wx);

  // 3) leaf gate biases
  leaf_bias<<<1, 192, 0, stream>>>(Ui, Uo, Uu, cb, bias3);

  // 4) leaf level (d = 13)
  leaves<<<(8192 * 64) / 256, 256, 0, stream>>>(Wx, bias3, h_all, c_all);

  // 5) inner levels d = 12 .. 0 (serial dependency)
  for (int d = DEPTH - 2; d >= 0; --d) {
    int n = 1 << d;
    int start = n - 1;
    int blocks = (n + NPB - 1) / NPB;
    tree_level<<<blocks, dim3(64, NPB), 0, stream>>>(
        Wx, Ui, Uf, Uo, Uu, cW, cb, h_all, c_all, start, n, out, d == 0);
  }
}
